// AttentionAverageStdScalingModule_85023172592502
// MI455X (gfx1250) — matrix operand order repaired and checked by reading, not compiled, not run on hardware
//
#include <hip/hip_runtime.h>
#include <hip/hip_bf16.h>

// ---------------------------------------------------------------------------
// AttentionAverageStdScaling fused pipeline for gfx1250 (MI455X, wave32).
//   K1  convert+transpose f32->bf16, fused L2-norm reduction  (one HBM pass)
//   K2  labels 88x88 -> 22x22 bilinear (exact 0.25*(2x2) taps)
//   K3  WMMA bf16 GEMM (cos-sim) + softmax + label attention, fused in LDS
//       -- full-tile B fragment preload (16x b128 clause, partial waits),
//          single buffer to stay under 256 VGPRs (no scratch spills)
//   K4  bilinear 22->88 upsample + cross-memory mean/var + certainty + add
// ---------------------------------------------------------------------------

#define NMEM 30
#define NSEQ 16
#define CCH  256
#define P2   484          // 22*22
#define PROW 512          // padded row count for bf16 panels
#define JPAD 496          // 31 tiles of 16 covering 484 (pad = -inf logits)
#define WL   88
#define ALPHA 20.0f

typedef __attribute__((ext_vector_type(16))) __bf16 v16bf;
typedef __attribute__((ext_vector_type(8)))  __bf16 bf16x8;
typedef __attribute__((ext_vector_type(8)))  float  v8f;

union FragBF {
    v16bf v;
    bf16x8 h[2];
};
union BF16Bits {
    unsigned short u;
    __bf16 b;
};

static __device__ inline __bf16 f2bf(float f) {
    unsigned u = __float_as_uint(f);
    u += 0x7FFFu + ((u >> 16) & 1u);          // round-to-nearest-even
    BF16Bits r; r.u = (unsigned short)(u >> 16);
    return r.b;
}

// ---------------------------------------------------------------------------
// K1: src [a][b][C][484] f32  ->  dst [swap(b,a)][PROW][C] bf16 (+ rsqrt norms)
// grid = (8 i-tiles of 64, a*b); block = 256
// ---------------------------------------------------------------------------
__global__ void k_convert_transpose(const float* __restrict__ src,
                                    __bf16* __restrict__ dst,
                                    float* __restrict__ inv,
                                    int a_dim, int b_dim) {
    __shared__ float tile[32][65];
    __shared__ float red[64][4];

    const int tid = threadIdx.x;
    const int o   = blockIdx.y;               // src outer index, [a][b] order
    const int ai  = o / b_dim;
    const int bi  = o % b_dim;
    const int dst_o = bi * a_dim + ai;        // dst outer index, [b][a] order
    const int i0  = blockIdx.x * 64;

    const float* sbase = src + (size_t)o * CCH * P2;
    __bf16* dbase = dst + (size_t)dst_o * PROW * CCH;

    float acc_sq = 0.0f;
    const int wi = tid >> 2;                  // 0..63  output row within tile
    const int g  = tid & 3;                   // 8-wide c group

    for (int c0 = 0; c0 < CCH; c0 += 32) {
        // load 32(c) x 64(i), coalesced along i
        #pragma unroll
        for (int e = 0; e < 8; ++e) {
            int idx = tid + e * 256;
            int cl = idx >> 6;
            int il = idx & 63;
            int ig = i0 + il;
            float v = (ig < P2) ? sbase[(size_t)(c0 + cl) * P2 + ig] : 0.0f;
            tile[cl][il] = v;
        }
        __syncthreads();
        // write 64(i) x 32(c), contiguous c per thread; accumulate sum(v^2)
        #pragma unroll
        for (int k = 0; k < 8; ++k) {
            int cl = g * 8 + k;
            float v = tile[cl][wi];
            acc_sq += v * v;
            dbase[(size_t)(i0 + wi) * CCH + c0 + cl] = f2bf(v);
        }
        __syncthreads();
    }

    red[wi][g] = acc_sq;
    __syncthreads();
    if (tid < 64) {
        float s = red[tid][0] + red[tid][1] + red[tid][2] + red[tid][3];
        int ig = i0 + tid;
        if (ig < PROW) inv[(size_t)dst_o * PROW + ig] = (s > 0.0f) ? rsqrtf(s) : 0.0f;
    }
}

// ---------------------------------------------------------------------------
// K2: labels (30,16,88,88) -> labels_down (m*16+n, 484)
// half-pixel bilinear at 4i+1.5 == 0.25 * 2x2 box at (4i+1 .. 4i+2)
// ---------------------------------------------------------------------------
__global__ void k_labels_down(const float* __restrict__ lab,
                              float* __restrict__ out) {
    const int o = blockIdx.x;                 // m*16+n
    const float* L = lab + (size_t)o * WL * WL;
    for (int i = threadIdx.x; i < P2; i += blockDim.x) {
        int wf = i / 22, hf = i % 22;
        int y = 4 * wf + 1, x = 4 * hf + 1;
        float v = 0.25f * (L[y * WL + x] + L[y * WL + x + 1] +
                           L[(y + 1) * WL + x] + L[(y + 1) * WL + x + 1]);
        out[(size_t)o * P2 + i] = v;
    }
}

// ---------------------------------------------------------------------------
// K3: fused cosine-sim GEMM (WMMA bf16) + softmax(axis=j) + label attention.
// grid = (31 i-tiles, nm=480);  block = 128 (4 waves of 32).
// A panel: te[n]  [PROW][256] bf16 ;  B panel: tr[n*30+m] [PROW][256] bf16
// All 16 b128 loads of a tile are issued before the 8-deep WMMA chain so the
// scheduler can use partial s_wait_loadcnt and overlap loads with WMMAs.
// ---------------------------------------------------------------------------
__global__ void __launch_bounds__(128)
k_sim_softmax_attn(const __bf16* __restrict__ A,
                   const __bf16* __restrict__ B,
                   const float* __restrict__ inv_te,
                   const float* __restrict__ inv_tr,
                   const float* __restrict__ labels_down,
                   const float* __restrict__ temp_ptr,
                   float* __restrict__ pmt) {
    __shared__ float S[16][JPAD];             // scaled logits
    __shared__ float labLDS[JPAD];
    __shared__ float irLDS[JPAD];
    __shared__ float itLDS[16];

    const int tid  = threadIdx.x;
    const int wave = tid >> 5;
    const int lane = tid & 31;

    const int i0 = blockIdx.x * 16;
    const int nm = blockIdx.y;
    const int m  = nm >> 4;                   // 0..29
    const int n  = nm & 15;                   // 0..15

    const float temp = temp_ptr[0];

    // stage per-row/col scale factors and labels in LDS
    if (tid < 16) itLDS[tid] = inv_te[(size_t)n * PROW + i0 + tid];
    for (int j = tid; j < JPAD; j += 128) {
        irLDS[j]  = inv_tr[((size_t)n * NMEM + m) * PROW + j];
        labLDS[j] = (j < P2) ? labels_down[((size_t)m * NSEQ + n) * P2 + j] : 0.0f;
    }
    __syncthreads();

    // A fragments (16 rows x K=256) held in registers; ISA 16-bit A layout:
    // lane l: row = l&15, VGPR0-3 <- K = koff..koff+7, VGPR4-7 <- K+16..K+23,
    // koff = (l>=16)*8, per 32-wide K step.
    const __bf16* Abase = A + ((size_t)n * PROW + i0 + (lane & 15)) * CCH;
    const int koffA = (lane >> 4) * 8;
    FragBF a[8];
    #pragma unroll
    for (int kk = 0; kk < 8; ++kk) {
        const __bf16* p = Abase + kk * 32 + koffA;
        a[kk].h[0] = *(const bf16x8*)(p);
        a[kk].h[1] = *(const bf16x8*)(p + 16);
    }

    // B layout: lane l holds column N=l&15, K = koff..koff+7, koff+16..koff+23, koff=(l>=16)*8
    const __bf16* Bpanel = B + ((size_t)n * NMEM + m) * PROW * CCH;
    const int col   = lane & 15;
    const int koffB = (lane >> 4) * 8;
    const int iLoc  = ((lane >> 4) << 3);

    for (int jt = wave; jt < 31; jt += 4) {
        const int j0 = jt * 16;

        // ---- preload the full 16x(K=256) B tile: one clause of 16 b128s ----
        FragBF b[8];
        const __bf16* base = Bpanel + (size_t)(j0 + col) * CCH + koffB;
        #pragma unroll
        for (int kk = 0; kk < 8; ++kk) {
            const __bf16* p = base + kk * 32;
            b[kk].h[0] = *(const bf16x8*)(p);
            b[kk].h[1] = *(const bf16x8*)(p + 16);
        }
        if (jt + 4 < 31)   // warm L0/L2 for this wave's next tile
            __builtin_prefetch(
                Bpanel + (size_t)((jt + 4) * 16 + col) * CCH + koffB, 0, 1);

        // ---- 8 chained WMMAs (partial loadcnt waits, no spills) ----
        v8f acc = {0.f, 0.f, 0.f, 0.f, 0.f, 0.f, 0.f, 0.f};
        #pragma unroll
        for (int kk = 0; kk < 8; ++kk)
            acc = __builtin_amdgcn_wmma_f32_16x16x32_bf16(
                false, a[kk].v, false, b[kk].v, (short)0, acc, false, false);

        // C/D layout: VGPR r -> M = r + 8*(lane>=16), N = lane&15
        const int jg = j0 + col;
        #pragma unroll
        for (int r = 0; r < 8; ++r) {
            int il = r + iLoc;
            float v = (jg < P2)
                    ? acc[r] * temp * itLDS[il] * irLDS[jg]
                    : -__builtin_inff();
            S[il][jg] = v;
        }
    }
    __syncthreads();

    // per-row softmax over 484 (pad = -inf) + dot with labels; 4 rows/wave
    for (int r = wave; r < 16; r += 4) {
        float mx = -__builtin_inff();
        for (int j = lane; j < JPAD; j += 32) mx = fmaxf(mx, S[r][j]);
        #pragma unroll
        for (int off = 16; off > 0; off >>= 1)
            mx = fmaxf(mx, __shfl_xor(mx, off, 32));

        float den = 0.0f, num = 0.0f;
        for (int j = lane; j < JPAD; j += 32) {
            float e = __expf(S[r][j] - mx);   // pad cols: exp(-inf)=0
            den += e;
            num += e * labLDS[j];
        }
        #pragma unroll
        for (int off = 16; off > 0; off >>= 1) {
            den += __shfl_xor(den, off, 32);
            num += __shfl_xor(num, off, 32);
        }
        int ig = i0 + r;
        if (lane == 0 && ig < P2)
            pmt[((size_t)m * NSEQ + n) * P2 + ig] = num / den;
    }
}

// ---------------------------------------------------------------------------
// K4: bilinear 22->88 per memory + mean / unbiased var over 30 memories,
//     certainty = exp(a/(1+var)-a), out = certainty*mean + test_scores
// ---------------------------------------------------------------------------
__global__ void k_finalize(const float* __restrict__ pmt,
                           const float* __restrict__ test_scores,
                           float* __restrict__ out) {
    const int idx = blockIdx.x * blockDim.x + threadIdx.x;   // 16*88*88
    const int n   = idx / (WL * WL);
    const int rem = idx % (WL * WL);
    const int y   = rem / WL;
    const int x   = rem % WL;

    float fy = 0.25f * (float)y - 0.375f;
    float fx = 0.25f * (float)x - 0.375f;
    float y0f = floorf(fy), x0f = floorf(fx);
    float wy = fy - y0f, wx = fx - x0f;
    int y0 = (int)y0f, x0 = (int)x0f;
    int ya = max(y0, 0),      xa = max(x0, 0);
    int yb = min(y0 + 1, 21), xb = min(x0 + 1, 21);
    float w00 = (1.f - wy) * (1.f - wx), w01 = (1.f - wy) * wx;
    float w10 = wy * (1.f - wx),         w11 = wy * wx;

    float s = 0.0f, s2 = 0.0f;
    #pragma unroll 5
    for (int m = 0; m < NMEM; ++m) {
        const float* p = pmt + ((size_t)m * NSEQ + n) * P2;
        float v = w00 * p[ya * 22 + xa] + w01 * p[ya * 22 + xb] +
                  w10 * p[yb * 22 + xa] + w11 * p[yb * 22 + xb];
        s  += v;
        s2 += v * v;
    }
    float mean = s * (1.0f / NMEM);
    float var  = (s2 - s * s * (1.0f / NMEM)) * (1.0f / (NMEM - 1));
    float cert = __expf(ALPHA / (1.0f + var) - ALPHA);
    out[idx] = cert * mean + test_scores[idx];
}

// ---------------------------------------------------------------------------

extern "C" void kernel_launch(void* const* d_in, const int* in_sizes, int n_in,
                              void* d_out, int out_size, void* d_ws, size_t ws_size,
                              hipStream_t stream) {
    const float* test_scores  = (const float*)d_in[0];  // (1,16,88,88)
    const float* train_labels = (const float*)d_in[1];  // (30,16,88,88)
    const float* test_feat    = (const float*)d_in[2];  // (1,16,256,22,22)
    const float* train_feats  = (const float*)d_in[3];  // (30,16,256,22,22)
    const float* softmax_temp = (const float*)d_in[4];  // (1,)
    float* out = (float*)d_out;

    // workspace carve-up (256B aligned)
    char* ws = (char*)d_ws;
    size_t off = 0;
    auto take = [&](size_t bytes) {
        void* p = ws + off;
        off += (bytes + 255) & ~(size_t)255;
        return p;
    };
    __bf16* Abf   = (__bf16*)take((size_t)NSEQ * PROW * CCH * sizeof(__bf16));
    __bf16* Bbf   = (__bf16*)take((size_t)NSEQ * NMEM * PROW * CCH * sizeof(__bf16));
    float* inv_te = (float*)take((size_t)NSEQ * PROW * sizeof(float));
    float* inv_tr = (float*)take((size_t)NSEQ * NMEM * PROW * sizeof(float));
    float* lab_dn = (float*)take((size_t)NMEM * NSEQ * P2 * sizeof(float));
    float* pmt    = (float*)take((size_t)NMEM * NSEQ * P2 * sizeof(float));
    (void)ws_size;

    // K1: convert + transpose + norms (test: a=1,b=16 ; train: a=30,b=16)
    k_convert_transpose<<<dim3(8, NSEQ), 256, 0, stream>>>(
        test_feat, Abf, inv_te, 1, NSEQ);
    k_convert_transpose<<<dim3(8, NMEM * NSEQ), 256, 0, stream>>>(
        train_feats, Bbf, inv_tr, NMEM, NSEQ);

    // K2: downsample labels
    k_labels_down<<<NMEM * NSEQ, 256, 0, stream>>>(train_labels, lab_dn);

    // K3: WMMA cosine-sim + softmax + attention
    k_sim_softmax_attn<<<dim3(31, NMEM * NSEQ), 128, 0, stream>>>(
        Abf, Bbf, inv_te, inv_tr, lab_dn, softmax_temp, pmt);

    // K4: upsample + mean/var + certainty + residual add
    k_finalize<<<dim3((NSEQ * WL * WL) / 256), 256, 0, stream>>>(
        pmt, test_scores, out);
}
